// ScaledDotProductAttention_58772332478987
// MI455X (gfx1250) — matrix operand order repaired and checked by reading, not compiled, not run on hardware
//
#include <hip/hip_runtime.h>

typedef __attribute__((ext_vector_type(16))) _Float16 v16h;
typedef __attribute__((ext_vector_type(8)))  _Float16 v8h;
typedef __attribute__((ext_vector_type(4)))  _Float16 v4h;
typedef __attribute__((ext_vector_type(2)))  _Float16 v2h;
typedef __attribute__((ext_vector_type(8)))  float    v8f;

#define B_ 4
#define H_ 16
#define S_ 2048
#define D_ 64

// 1/sqrt(64) * log2(e): scores come out of QK^T already in the exp2 domain.
static constexpr float Q_SCALE = 0.125f * 1.44269504088896340736f;

// Workspace layout (bytes)
static constexpr size_t KH_OFF = 0;                                      // f16 K, [bh][key][d]
static constexpr size_t VT_OFF = (size_t)B_ * H_ * S_ * D_ * 2;          // f16 V^T, [bh][d][key]
static constexpr size_t MH_OFF = VT_OFF + (size_t)B_ * H_ * S_ * D_ * 2; // f16 mult-mask [b][q][k]

// ---------------- Preprocess 1: K f32 -> f16 (same layout) ----------------
__global__ __launch_bounds__(256) void cvt_k_kernel(const float* __restrict__ K,
                                                    _Float16* __restrict__ Kh) {
  size_t i = ((size_t)blockIdx.x * 256 + threadIdx.x) * 4;
  const float4 f = *(const float4*)&K[i];
  v4h o = { (_Float16)f.x, (_Float16)f.y, (_Float16)f.z, (_Float16)f.w };
  *(v4h*)&Kh[i] = o;
}

// ---------------- Preprocess 2: V f32 [key][d] -> f16 V^T [d][key] ----------------
__global__ __launch_bounds__(256) void transpose_v_kernel(const float* __restrict__ V,
                                                          _Float16* __restrict__ Vt) {
  __shared__ _Float16 tile[16][72];          // 16 keys x 64 d, padded rows
  const int blk  = blockIdx.x;               // bh*128 + key-tile
  const int bh   = blk >> 7;
  const int key0 = (blk & 127) << 4;
  const float* v = V + (size_t)bh * S_ * D_;
  _Float16*   vt = Vt + (size_t)bh * D_ * S_;
  const int t = threadIdx.x;

  {
    const int ky = t >> 4;
    const int d4 = (t & 15) * 4;
    const float4 f = *(const float4*)&v[(size_t)(key0 + ky) * D_ + d4];
    tile[ky][d4 + 0] = (_Float16)f.x;
    tile[ky][d4 + 1] = (_Float16)f.y;
    tile[ky][d4 + 2] = (_Float16)f.z;
    tile[ky][d4 + 3] = (_Float16)f.w;
  }
  __syncthreads();
  {
    const int d  = t >> 2;
    const int j4 = (t & 3) * 4;
    v4h o = { tile[j4 + 0][d], tile[j4 + 1][d], tile[j4 + 2][d], tile[j4 + 3][d] };
    *(v4h*)&vt[(size_t)d * S_ + key0 + j4] = o;
  }
}

// ---------------- Preprocess 3: int32 mask -> multiplicative f16 (mask==1 -> 0, else 1) ----
__global__ __launch_bounds__(256) void mask_f16_kernel(const int* __restrict__ Mask,
                                                       _Float16* __restrict__ Mh) {
  size_t i = ((size_t)blockIdx.x * 256 + threadIdx.x) * 4;
  const int4 mv = *(const int4*)&Mask[i];
  v4h o = { (_Float16)(mv.x == 1 ? 0.0f : 1.0f),
            (_Float16)(mv.y == 1 ? 0.0f : 1.0f),
            (_Float16)(mv.z == 1 ? 0.0f : 1.0f),
            (_Float16)(mv.w == 1 ? 0.0f : 1.0f) };
  *(v4h*)&Mh[i] = o;
}

// ---------------- Main flash-attention kernel ----------------
// One wave per 32-row Q chunk (two 16x64 A tiles); 8 waves per block.
// K/V B-tiles are loaded once per iteration and shared by both Q tiles.
// All streaming loads use one lane-fixed base + (kb + compile-time imm) so the
// compiler keeps a single address pair per stream (stays under 256 VGPRs).
__global__ __launch_bounds__(256) void fa_fwd_kernel(
    const float* __restrict__ Q, const _Float16* __restrict__ KH,
    const _Float16* __restrict__ VT, const _Float16* __restrict__ MH,
    float* __restrict__ Out) {
  __shared__ __align__(16) _Float16 ldsP[8 * 2 * 16 * 32];  // 16 KB: per-wave 2x(16x32) P staging

  const int lane = threadIdx.x & 31;
  const int wave = threadIdx.x >> 5;
  const int hl   = lane >> 4;      // 16-lane half
  const int ln16 = lane & 15;

  const int job = blockIdx.x * 8 + wave;    // 0..4095
  const int bh  = job >> 6;                 // job / (S/32)
  const int q0  = (job & 63) << 5;          // 32-row chunk
  const int b   = bh / H_;

  const float*    q  = Q  + (size_t)bh * S_ * D_;
  const _Float16* kh = KH + (size_t)bh * S_ * D_;
  const _Float16* vt = VT + (size_t)bh * D_ * S_;
  const _Float16* mh = MH + (size_t)b * S_ * S_;
  float*          o  = Out + (size_t)bh * S_ * D_;

  // Lane-fixed stream bases: loop index contributes only kb + constant imm.
  const _Float16* krow = kh + (size_t)(2 * ln16) * D_ + 16 * hl;        // key = kb + 2*ln16 (+1)
  const _Float16* vrow = vt + (size_t)ln16 * S_ + 16 * hl;              // d col base
  const _Float16* mrow = mh + (size_t)(q0 + 8 * hl) * S_ + 2 * ln16;    // mask row base

  _Float16* myP = &ldsP[wave * (2 * 16 * 32)];
  const int m = ln16;              // A-matrix row owned by this lane

  // ---- Q: two q-tiles x two 16x32 f16 A-matrices, pre-scaled into exp2 domain ----
  v16h aQ[2][2];
  #pragma unroll
  for (int qt = 0; qt < 2; ++qt) {
    #pragma unroll
    for (int c = 0; c < 2; ++c) {
      #pragma unroll
      for (int vv = 0; vv < 8; ++vv) {
        int d = c * 32 + 16 * hl + 2 * vv;
        const float2 f = *(const float2*)&q[(size_t)(q0 + qt * 16 + m) * D_ + d];
        aQ[qt][c][2 * vv]     = (_Float16)(f.x * Q_SCALE);
        aQ[qt][c][2 * vv + 1] = (_Float16)(f.y * Q_SCALE);
      }
    }
  }

  // B tile of ones in column 0 only -> row-sum accumulator via WMMA
  v16h bOnes;
  {
    _Float16 one0 = (ln16 == 0) ? (_Float16)1.0f : (_Float16)0.0f;
    #pragma unroll
    for (int e = 0; e < 16; ++e) bOnes[e] = one0;
  }

  const v8f vzero = {};
  v8f acc[2][4];
  #pragma unroll
  for (int qt = 0; qt < 2; ++qt)
    #pragma unroll
    for (int c = 0; c < 4; ++c) acc[qt][c] = vzero;
  v8f accL[2] = { vzero, vzero };

  for (int kb = 0; kb < S_; kb += 32) {
    // ---- scores: tile t, column n <-> key kb + 2n + t; shared bK for both q-tiles ----
    v8f s[2][2];     // [qt][t]
    #pragma unroll
    for (int qt = 0; qt < 2; ++qt) { s[qt][0] = vzero; s[qt][1] = vzero; }
    #pragma unroll
    for (int c = 0; c < 2; ++c) {
      const v16h bK0 = *(const v16h*)&krow[(size_t)kb * D_ + c * 32];        // key even
      const v16h bK1 = *(const v16h*)&krow[(size_t)kb * D_ + D_ + c * 32];   // key odd
      #pragma unroll
      for (int qt = 0; qt < 2; ++qt) {
        s[qt][0] = __builtin_amdgcn_wmma_f32_16x16x32_f16(false, aQ[qt][c], false, bK0, (short)0, s[qt][0], false, false);
        s[qt][1] = __builtin_amdgcn_wmma_f32_16x16x32_f16(false, aQ[qt][c], false, bK1, (short)0, s[qt][1], false, false);
      }
    }

    // ---- unnormalized exp2 (raw v_exp_f32) + f16 mask + stage P pairs to LDS ----
    #pragma unroll
    for (int qt = 0; qt < 2; ++qt) {
      #pragma unroll
      for (int r = 0; r < 8; ++r) {
        const v2h mp = *(const v2h*)&mrow[(size_t)(qt * 16 + r) * S_ + kb];
        const float e0 = __builtin_amdgcn_exp2f(s[qt][0][r]);
        const float e1 = __builtin_amdgcn_exp2f(s[qt][1][r]);
        v2h pk; pk[0] = (_Float16)e0; pk[1] = (_Float16)e1;  // v_cvt_pk_f16_f32
        pk = pk * mp;                                        // v_pk_mul_f16
        *(v2h*)&myP[qt * 512 + (r + 8 * hl) * 32 + 2 * ln16] = pk;
      }
    }

    // ---- reload both P tiles in A layout ----
    v16h aP[2];
    #pragma unroll
    for (int qt = 0; qt < 2; ++qt) {
      const v8h lo = *(const v8h*)&myP[qt * 512 + m * 32 + 16 * hl];
      const v8h hi = *(const v8h*)&myP[qt * 512 + m * 32 + 16 * hl + 8];
      aP[qt] = __builtin_shufflevector(lo, hi, 0, 1, 2, 3, 4, 5, 6, 7,
                                       8, 9, 10, 11, 12, 13, 14, 15);
    }

    // ---- row sums and O += P x V (bV shared by both q-tiles) ----
    #pragma unroll
    for (int qt = 0; qt < 2; ++qt)
      accL[qt] = __builtin_amdgcn_wmma_f32_16x16x32_f16(false, aP[qt], false, bOnes, (short)0, accL[qt], false, false);
    #pragma unroll
    for (int c = 0; c < 4; ++c) {
      const v16h bV = *(const v16h*)&vrow[(size_t)(c * 16) * S_ + kb];
      #pragma unroll
      for (int qt = 0; qt < 2; ++qt)
        acc[qt][c] = __builtin_amdgcn_wmma_f32_16x16x32_f16(false, aP[qt], false, bV, (short)0, acc[qt][c], false, false);
    }
  }

  // ---- epilogue: normalize by row sums (col 0 of accL) and store ----
  #pragma unroll
  for (int qt = 0; qt < 2; ++qt) {
    #pragma unroll
    for (int r = 0; r < 8; ++r) {
      const float Ls  = __shfl(accL[qt][r], 16 * hl, 32);  // row sum for row r+8*hl
      const float inv = __builtin_amdgcn_rcpf(Ls);
      const int row = q0 + qt * 16 + r + 8 * hl;
      #pragma unroll
      for (int c = 0; c < 4; ++c)
        o[(size_t)row * D_ + c * 16 + ln16] = acc[qt][c][r] * inv;
    }
  }
}

extern "C" void kernel_launch(void* const* d_in, const int* in_sizes, int n_in,
                              void* d_out, int out_size, void* d_ws, size_t ws_size,
                              hipStream_t stream) {
  (void)in_sizes; (void)n_in; (void)out_size; (void)ws_size;
  const float* q    = (const float*)d_in[0];
  const float* k    = (const float*)d_in[1];
  const float* v    = (const float*)d_in[2];
  const int*   mask = (const int*)d_in[3];
  float*       out  = (float*)d_out;

  char* ws = (char*)d_ws;
  _Float16* Kh = (_Float16*)(ws + KH_OFF);
  _Float16* Vt = (_Float16*)(ws + VT_OFF);
  _Float16* Mh = (_Float16*)(ws + MH_OFF);

  // Preprocess: K->f16, V->f16 transposed, mask->multiplicative f16
  {
    const size_t kelems = (size_t)B_ * H_ * S_ * D_;       // 8.39M
    cvt_k_kernel<<<dim3((unsigned)(kelems / (256 * 4))), dim3(256), 0, stream>>>(k, Kh);
    transpose_v_kernel<<<dim3(B_ * H_ * (S_ / 16)), dim3(256), 0, stream>>>(v, Vt);
    const size_t melems = (size_t)B_ * S_ * S_;            // 16.8M
    mask_f16_kernel<<<dim3((unsigned)(melems / (256 * 4))), dim3(256), 0, stream>>>(mask, Mh);
  }

  // Main: 4096 32-row Q chunks, 8 waves per block
  fa_fwd_kernel<<<dim3(B_ * H_ * (S_ / 32) / 8), dim3(256), 0, stream>>>(q, Kh, Vt, Mh, out);
}
